// RNN_54769422959080
// MI455X (gfx1250) — matrix operand order repaired and checked by reading, not compiled, not run on hardware
//
#include <hip/hip_runtime.h>
#include <math.h>

#define SEQ   512
#define BATCH 64
#define DIN   512
#define HID   1024
#define DOUT  512
#define KC    1536   // K of both GEMMs (DIN+HID == DOUT+HID)
#define N1    1536   // GEMM1 output cols: [out(512) | h_new(1024)]
#define NKB   48     // K blocks of 32

#define NWG   72
#define TPB   256
#define WPG   8      // wave32 waves per workgroup
#define G1_WAVES 384 // 4 m-tiles x 96 n-tiles  (WGs 0..47)
#define G2_WAVES 128 // 4 m-tiles x 32 n-tiles  (WGs 48..63)
                     // softmax: 64 waves       (WGs 64..71)

// LDS: per WG two N-tile weight slabs, each stored in WMMA register order:
// slab[kb][lane][16 u16]  -> 48 * 32 * 16 = 24576 u16 = 48KB per slab, 96KB total
#define SLAB_U16 (NKB * 32 * 16)

typedef __attribute__((ext_vector_type(16))) __bf16 v16bf;
typedef __attribute__((ext_vector_type(8)))  __bf16 v8bf;
typedef __attribute__((ext_vector_type(8)))  float  v8f;

// float -> bf16 (round-to-nearest-even), kept as u16 to avoid __bf16 scalar math
static __device__ __forceinline__ unsigned short f2bf(float f) {
  unsigned u = __float_as_uint(f);
  u += 0x7fffu + ((u >> 16) & 1u);
  return (unsigned short)(u >> 16);
}

// ---------------- one-time prep kernels ----------------

__global__ void k_cvt(const float* __restrict__ src, unsigned short* __restrict__ dst, int n) {
  for (int i = blockIdx.x * blockDim.x + threadIdx.x; i < n; i += gridDim.x * blockDim.x)
    dst[i] = f2bf(src[i]);
}

// Wc rows 0..511 = W_i2o, rows 512..1535 = W_i2h (both [*, 1536] row-major)
__global__ void k_pack_wc(const float* __restrict__ Wi2o, const float* __restrict__ Wi2h,
                          unsigned short* __restrict__ Wc) {
  const int n = N1 * KC;
  for (int i = blockIdx.x * blockDim.x + threadIdx.x; i < n; i += gridDim.x * blockDim.x) {
    int row = i / KC, col = i - row * KC;
    float v = (row < DOUT) ? Wi2o[(size_t)row * KC + col]
                           : Wi2h[(size_t)(row - DOUT) * KC + col];
    Wc[i] = f2bf(v);
  }
}

__global__ void k_pack_bias(const float* __restrict__ bi2o, const float* __restrict__ bi2h,
                            const float* __restrict__ bo2o,
                            float* __restrict__ bc, float* __restrict__ bo) {
  int i = blockIdx.x * blockDim.x + threadIdx.x;
  if (i < DOUT)             bc[i] = bi2o[i];
  else if (i < N1)          bc[i] = bi2h[i - DOUT];
  else if (i < N1 + DOUT)   bo[i - N1] = bo2o[i - N1];
}

// ---------------- WMMA K-span: A from global, B from LDS ----------------
// A lane (row = lane&15): chunks at K = kb*32 + 8*half and +16  (two 16B global loads)
// B from LDS in register order: 32B contiguous per lane, conflict-free ds_load_b128
static __device__ __forceinline__ v8f mm_span_lds(v8f acc,
                                                  const unsigned short* __restrict__ pa,
                                                  const unsigned short* lb,
                                                  int nk) {
#pragma unroll 4
  for (int k = 0; k < nk; ++k) {
    v8bf a0 = *(const v8bf*)(pa);
    v8bf a1 = *(const v8bf*)(pa + 16);
    v8bf b0 = *(const v8bf*)(lb);
    v8bf b1 = *(const v8bf*)(lb + 8);
    __builtin_prefetch(pa + 64, 0, 1);  // next A chunks -> global_prefetch_b8
    v16bf a = __builtin_shufflevector(a0, a1, 0,1,2,3,4,5,6,7,8,9,10,11,12,13,14,15);
    v16bf b = __builtin_shufflevector(b0, b1, 0,1,2,3,4,5,6,7,8,9,10,11,12,13,14,15);
    acc = __builtin_amdgcn_wmma_f32_16x16x32_bf16(false, a, false, b, (short)0, acc,
                                                  false, false);
    pa += 32;
    lb += 512;  // next kb slab
  }
  return acc;
}

// ---------------- persistent pipelined RNN kernel ----------------

__global__ __launch_bounds__(TPB, 1) void k_rnn(
    const unsigned short* __restrict__ xb,   // [SEQ][BATCH][DIN] bf16
    const float* __restrict__ mask,          // [SEQ][BATCH][DOUT] f32
    const unsigned short* __restrict__ Wc,   // [N1][KC] bf16
    const unsigned short* __restrict__ Wo,   // [DOUT][KC] bf16
    const float* __restrict__ bc,            // [N1]
    const float* __restrict__ bo,            // [DOUT]
    unsigned short* __restrict__ zbuf,       // [2][BATCH][N1] bf16 (zeroed)
    float* __restrict__ ocbuf,               // [2][BATCH][DOUT]
    float* __restrict__ out,                 // [SEQ][BATCH][DOUT]
    unsigned* __restrict__ bar)              // [0]=counter [1]=flag (zeroed)
{
  extern __shared__ unsigned short lds_w[];  // 2 slabs x 48KB

  const int wid  = threadIdx.x >> 5;
  const int wave = blockIdx.x * WPG + wid;
  const int lane = threadIdx.x & 31;
  const int half = lane >> 4;
  const int rc   = lane & 15;
  const int nt   = wid >> 2;                 // WG-local N-tile (0/1)

  // ---- one-time: stage this WG's weight slabs into LDS (register order) ----
  if (wave < G1_WAVES + G2_WAVES) {
    if ((wid & 3) == 0) {                    // one wave per slab
      const bool g2 = (wave >= G1_WAVES);
      const int n0  = g2 ? (((wave - G1_WAVES) >> 2) * 16) : ((wave >> 2) * 16);
      const unsigned short* gsrc =
          (g2 ? Wo : Wc) + (size_t)(n0 + rc) * KC + half * 8;
      unsigned short* ldst = lds_w + nt * SLAB_U16 + lane * 16;
#pragma unroll 4
      for (int kb = 0; kb < NKB; ++kb) {
        v8bf c0 = *(const v8bf*)(gsrc);
        v8bf c1 = *(const v8bf*)(gsrc + 16);
        *(v8bf*)(ldst) = c0;
        *(v8bf*)(ldst + 8) = c1;
        gsrc += 32;
        ldst += 512;
      }
    }
  }
  __syncthreads();

  const unsigned short* lbase = lds_w + nt * SLAB_U16 + lane * 16;

  for (int p = 0; p < SEQ + 2; ++p) {
    if (wave < G1_WAVES) {
      // ---- GEMM1 step t=p: z_t = [x_t | h_{t-1}] @ Wc^T + bc ----
      const int t = p;
      if (t < SEQ) {
        const int m0 = (wave & 3) * 16;
        const int n0 = (wave >> 2) * 16;
        const unsigned short* zprev = zbuf + (size_t)((t + 1) & 1) * BATCH * N1;
        unsigned short* zcur        = zbuf + (size_t)(t & 1) * BATCH * N1;
        v8f acc;
        const float bias = bc[n0 + rc];
#pragma unroll
        for (int r = 0; r < 8; ++r) acc[r] = bias;
        // K 0..511 from x_t
        const unsigned short* pax = xb + ((size_t)t * BATCH + (m0 + rc)) * DIN + half * 8;
        acc = mm_span_lds(acc, pax, lbase, 16);
        // K 512..1535 from h_{t-1} = z_{t-1}[:, 512:]
        const unsigned short* paz = zprev + (size_t)(m0 + rc) * N1 + DIN + half * 8;
        acc = mm_span_lds(acc, paz, lbase + 16 * 512, 32);
#pragma unroll
        for (int r = 0; r < 8; ++r)
          zcur[(size_t)(m0 + r + 8 * half) * N1 + (n0 + rc)] = f2bf(acc[r]);
      }
    } else if (wave < G1_WAVES + G2_WAVES) {
      // ---- GEMM2 step t=p-1: oc_t = z_t @ Wo^T + bo, then dropout ----
      const int t = p - 1;
      if (t >= 0 && t < SEQ) {
        const int id = wave - G1_WAVES;
        const int m0 = (id & 3) * 16;
        const int n0 = (id >> 2) * 16;
        const unsigned short* z = zbuf + (size_t)(t & 1) * BATCH * N1;
        float* oc               = ocbuf + (size_t)(t & 1) * BATCH * DOUT;
        v8f acc;
        const float bias = bo[n0 + rc];
#pragma unroll
        for (int r = 0; r < 8; ++r) acc[r] = bias;
        const unsigned short* pa = z + (size_t)(m0 + rc) * N1 + half * 8;
        acc = mm_span_lds(acc, pa, lbase, NKB);
#pragma unroll
        for (int r = 0; r < 8; ++r) {
          const int m = m0 + r + 8 * half;
          const float mv = mask[((size_t)t * BATCH + m) * DOUT + (n0 + rc)];
          oc[(size_t)m * DOUT + (n0 + rc)] = acc[r] * mv;
        }
      }
    } else {
      // ---- log_softmax step t=p-2, one wave per batch row ----
      const int t = p - 2;
      if (t >= 0) {
        const int b = wave - (G1_WAVES + G2_WAVES);
        const float* src = ocbuf + (size_t)(t & 1) * BATCH * DOUT + (size_t)b * DOUT + lane * 16;
        float v[16];
        const float4* p4 = (const float4*)src;
#pragma unroll
        for (int i = 0; i < 4; ++i) {
          float4 q = p4[i];
          v[4 * i + 0] = q.x; v[4 * i + 1] = q.y; v[4 * i + 2] = q.z; v[4 * i + 3] = q.w;
        }
        float mx = v[0];
#pragma unroll
        for (int i = 1; i < 16; ++i) mx = fmaxf(mx, v[i]);
#pragma unroll
        for (int off = 16; off > 0; off >>= 1) mx = fmaxf(mx, __shfl_xor(mx, off, 32));
        float s = 0.f;
#pragma unroll
        for (int i = 0; i < 16; ++i) s += __expf(v[i] - mx);
#pragma unroll
        for (int off = 16; off > 0; off >>= 1) s += __shfl_xor(s, off, 32);
        const float lse = mx + __logf(s);
        float* dst = out + ((size_t)t * BATCH + b) * DOUT + lane * 16;
#pragma unroll
        for (int i = 0; i < 4; ++i) {
          float4 q;
          q.x = v[4 * i + 0] - lse; q.y = v[4 * i + 1] - lse;
          q.z = v[4 * i + 2] - lse; q.w = v[4 * i + 3] - lse;
          ((float4*)dst)[i] = q;
        }
      }
    }

    // ---- device-wide barrier (monotonic flag, one per phase) ----
    __syncthreads();
    if (threadIdx.x == 0) {
      __threadfence();
      unsigned prev = atomicAdd(bar, 1u);
      if (prev == NWG - 1) {
        atomicExch(bar, 0u);                       // reset before release
        __threadfence();
        atomicExch(bar + 1, (unsigned)(p + 1));    // release
      } else {
        while (*(volatile unsigned*)(bar + 1) < (unsigned)(p + 1))
          __builtin_amdgcn_s_sleep(2);
      }
      __threadfence();
    }
    __syncthreads();
  }
}

// ---------------- host launch ----------------

extern "C" void kernel_launch(void* const* d_in, const int* in_sizes, int n_in,
                              void* d_out, int out_size, void* d_ws, size_t ws_size,
                              hipStream_t stream) {
  (void)in_sizes; (void)n_in; (void)out_size; (void)ws_size;
  const float* x     = (const float*)d_in[0];
  const float* mask  = (const float*)d_in[1];
  const float* Wi2h  = (const float*)d_in[2];
  const float* bi2h  = (const float*)d_in[3];
  const float* Wi2o  = (const float*)d_in[4];
  const float* bi2o  = (const float*)d_in[5];
  const float* Wo2o  = (const float*)d_in[6];
  const float* bo2o  = (const float*)d_in[7];
  float* out = (float*)d_out;

  char* ws = (char*)d_ws;
  size_t off = 0;
  auto take = [&](size_t bytes) -> char* {
    char* p = ws + off;
    off += (bytes + 255) & ~(size_t)255;
    return p;
  };
  unsigned*       bar  = (unsigned*)take(256);
  unsigned short* xb   = (unsigned short*)take((size_t)SEQ * BATCH * DIN * 2);
  unsigned short* Wc   = (unsigned short*)take((size_t)N1 * KC * 2);
  unsigned short* Wo   = (unsigned short*)take((size_t)DOUT * KC * 2);
  float*          bc   = (float*)take((size_t)N1 * 4);
  float*          bo   = (float*)take((size_t)DOUT * 4);
  unsigned short* zbuf = (unsigned short*)take((size_t)2 * BATCH * N1 * 2);
  float*          oc   = (float*)take((size_t)2 * BATCH * DOUT * 4);

  hipMemsetAsync(bar, 0, 256, stream);                               // barrier state
  hipMemsetAsync(zbuf, 0, (size_t)2 * BATCH * N1 * 2, stream);       // h_{-1} = 0

  k_cvt<<<2048, 256, 0, stream>>>(x, xb, SEQ * BATCH * DIN);
  k_pack_wc<<<2048, 256, 0, stream>>>(Wi2o, Wi2h, Wc);
  k_cvt<<<1024, 256, 0, stream>>>(Wo2o, Wo, DOUT * KC);
  k_pack_bias<<<8, 256, 0, stream>>>(bi2o, bi2h, bo2o, bc, bo);

  const size_t lds_bytes = (size_t)2 * SLAB_U16 * sizeof(unsigned short);  // 96KB
  k_rnn<<<NWG, TPB, lds_bytes, stream>>>(xb, mask, Wc, Wo, bc, bo, zbuf, oc, out, bar);
}
